// PoincareMLP_67448166417267
// MI455X (gfx1250) — matrix operand order repaired and checked by reading, not compiled, not run on hardware
//
#include <hip/hip_runtime.h>
#include <hip/hip_bf16.h>

// ---------------------------------------------------------------------------
// Poincare-ball MLP, fused per-layer GEMM + hyperbolic epilogue for gfx1250.
//   layer k:  acc = A(bf16) x B(bf16) via v_wmma_f32_16x16x32_bf16
//             w   = sinh(2*zn*asinh(cs*lam*acc/zn*cosh(tcr)-(lam-1)*sinh(tcr)))/cs
//             out = s_row * relu(w)   (s_row from per-row sums, LDS reduction)
//   row sumsq is forwarded between layers so lam never needs a re-reduction.
// Workspace layout (total ~114 MB, assumed <= ws_size):
//   row-sumsq ping/pong, bf16 transposed weights, per-col (zn,cosh,sinh),
//   bf16 intermediate activations h1/h2/h3.
// ---------------------------------------------------------------------------

typedef __bf16 bf16_t;
typedef __attribute__((ext_vector_type(16))) __bf16 v16bf;
typedef __attribute__((ext_vector_type(8)))  __bf16 v8bf;
typedef __attribute__((ext_vector_type(8)))  float  v8f;

#define CCURV 2.3026f
#define CS    1.5174321f      // sqrt(2.3026)
#define EPSF  1e-15f

// ---------------------------------------------------------------------------
// Per-row sum of squares of x (layer-1 conformal factor input).
// One wave per row, coalesced stride-32 reads, wave32 xor-shuffle reduce.
// ---------------------------------------------------------------------------
__global__ __launch_bounds__(256)
void row_sumsq_kernel(const float* __restrict__ x, float* __restrict__ ss, int D)
{
    const int gtid = blockIdx.x * blockDim.x + threadIdx.x;
    const int row  = gtid >> 5;
    const int lane = gtid & 31;
    const float* r = x + (long)row * D;
    float s = 0.f;
    for (int i = lane; i < D; i += 32) { float v = r[i]; s += v * v; }
    #pragma unroll
    for (int o = 16; o > 0; o >>= 1) s += __shfl_xor(s, o, 32);
    if (lane == 0) ss[row] = s;
}

// ---------------------------------------------------------------------------
// Weight prep: z (FI x FO, fp32 row-major) -> zT (FO_PAD x FI, bf16), plus
// per-column params (zn, cosh(2*cs*r), sinh(2*cs*r)). Pad columns are zeroed.
// ---------------------------------------------------------------------------
template<int FI, int FO, int FO_PAD>
__global__ __launch_bounds__(64)
void prep_weights_kernel(const float* __restrict__ z, const float* __restrict__ r,
                         bf16_t* __restrict__ zT, float* __restrict__ colp)
{
    const int n = blockIdx.x * blockDim.x + threadIdx.x;
    if (n >= FO_PAD) return;
    float ss = 0.f;
    if (n < FO) {
        for (int k = 0; k < FI; ++k) {
            float v = z[(long)k * FO + n];
            ss += v * v;
            zT[(long)n * FI + k] = (bf16_t)v;
        }
        float zn  = fmaxf(sqrtf(ss), EPSF);
        float tcr = 2.f * CS * r[n];
        colp[3 * n + 0] = zn;
        colp[3 * n + 1] = coshf(tcr);
        colp[3 * n + 2] = sinhf(tcr);
    } else {
        for (int k = 0; k < FI; ++k) zT[(long)n * FI + k] = (bf16_t)0.f;
        colp[3 * n + 0] = 1.f;  // benign pad params (never stored/reduced)
        colp[3 * n + 1] = 1.f;
        colp[3 * n + 2] = 0.f;
    }
}

// ---------------------------------------------------------------------------
// Fused hyperbolic layer.
//   grid.x workgroups of 32*NWM*NWN threads; each WG owns MT=16*NWM rows and
//   ALL FO_PAD output columns (waves tiled NWM x NWN), so the per-row
//   reductions needed by the ball projection / hrelu stay in one workgroup.
//   A fragments: loaded per-lane straight from global per the CDNA5 16-bit
//   A layout (lane half h holds K = [8h..8h+7] and [16+8h..16+8h+7]).
//   B fragments: two 16B loads per lane (K = 8h..8h+7, 16+8h..16+8h+7) from the bf16-transposed weights.
// ---------------------------------------------------------------------------
template<int FI, int FO_PAD, int NWM, int NWN, bool IN_BF16, bool LAST>
__global__ __launch_bounds__(32 * NWM * NWN)
void hlayer_kernel(const float*  __restrict__ inf,
                   const bf16_t* __restrict__ inb,
                   const float*  __restrict__ in_ss,
                   const bf16_t* __restrict__ zT,
                   const float*  __restrict__ colp,
                   bf16_t*       __restrict__ out_bf16,
                   float*        __restrict__ out_ss,
                   float*        __restrict__ out_f32)
{
    constexpr int MT    = 16 * NWM;
    constexpr int NFRAG = FO_PAD / (16 * NWN);

    const int tid   = threadIdx.x;
    const int wave  = tid >> 5;
    const int lane  = tid & 31;
    const int h     = lane >> 4;      // lane half (selects K sub-range)
    const int l16   = lane & 15;      // row (A) / column (B,C,D) within tile
    const int waveM = wave % NWM;
    const int waveN = wave / NWM;
    const long rowWG = (long)blockIdx.x * MT;
    const long rowW  = rowWG + waveM * 16;

    __shared__ float s_lam[MT];
    __shared__ float s_w2[MT];
    __shared__ float s_r2[MT];
    __shared__ float s_scale[MT];

    if (tid < MT) {
        s_w2[tid] = 0.f;
        s_r2[tid] = 0.f;
        s_lam[tid] = 2.f / (1.f - CCURV * in_ss[rowWG + tid]);
    }
    __syncthreads();

    v8f acc[NFRAG] = {};

    const long arow = (rowW + l16) * (long)FI;

    for (int k0 = 0; k0 < FI; k0 += 32) {
        // ---- A fragment (16x32 bf16, this wave's 16 rows) ----
        v16bf a;
        if (IN_BF16) {
            const bf16_t* ap = inb + arow + k0;
            v8bf alo = *(const v8bf*)(ap + 8 * h);         // K = 8h .. 8h+7
            v8bf ahi = *(const v8bf*)(ap + 16 + 8 * h);    // K = 16+8h .. +7
            #pragma unroll
            for (int j = 0; j < 8; ++j) { a[j] = alo[j]; a[8 + j] = ahi[j]; }
        } else {
            const float* ap = inf + arow + k0;
            float t[16];
            *(float4*)(t + 0)  = *(const float4*)(ap + 8 * h);
            *(float4*)(t + 4)  = *(const float4*)(ap + 8 * h + 4);
            *(float4*)(t + 8)  = *(const float4*)(ap + 16 + 8 * h);
            *(float4*)(t + 12) = *(const float4*)(ap + 16 + 8 * h + 4);
            #pragma unroll
            for (int j = 0; j < 16; ++j) a[j] = (bf16_t)t[j];
        }
        // ---- B fragments + WMMA over this wave's N blocks ----
        #pragma unroll
        for (int f = 0; f < NFRAG; ++f) {
            const int ncol = waveN * 16 * NFRAG + f * 16 + l16;
            const bf16_t* bp = zT + (long)ncol * FI + k0;
            v8bf blo = *(const v8bf*)(bp + 8 * h);
            v8bf bhi = *(const v8bf*)(bp + 16 + 8 * h);
            v16bf b;
            #pragma unroll
            for (int j = 0; j < 8; ++j) { b[j] = blo[j]; b[8 + j] = bhi[j]; }
            acc[f] = __builtin_amdgcn_wmma_f32_16x16x32_bf16(
                false, a, false, b, (short)0, acc[f], false, false);
        }
    }

    // ---- hyperbolic epilogue: acc -> w (signed hyperplane -> sinh map) ----
    #pragma unroll
    for (int f = 0; f < NFRAG; ++f) {
        const int ncol = waveN * 16 * NFRAG + f * 16 + l16;
        const float zn = colp[3 * ncol + 0];
        const float cT = colp[3 * ncol + 1];
        const float sT = colp[3 * ncol + 2];
        #pragma unroll
        for (int r = 0; r < 8; ++r) {
            const float lam = s_lam[waveM * 16 + r + 8 * h];
            const float tt  = CS * lam * acc[f][r] / zn * cT - (lam - 1.f) * sT;
            const float as  = __logf(tt + sqrtf(tt * tt + 1.f));   // asinh
            const float u   = 2.f * zn * as;                        // cs*d
            acc[f][r] = (__expf(u) - __expf(-u)) * (0.5f / CS);     // sinh/cs
        }
    }

    if (!LAST) {
        // per-lane partial row sums, then LDS ds_add_f32 across the 16 lanes
        // of each half and across the NWN column-waves.
        float rw2[8], rr2[8];
        #pragma unroll
        for (int r = 0; r < 8; ++r) { rw2[r] = 0.f; rr2[r] = 0.f; }
        #pragma unroll
        for (int f = 0; f < NFRAG; ++f)
            #pragma unroll
            for (int r = 0; r < 8; ++r) {
                const float w  = acc[f][r];
                const float wr = fmaxf(w, 0.f);
                rw2[r] += w * w;
                rr2[r] += wr * wr;
            }
        #pragma unroll
        for (int r = 0; r < 8; ++r) {
            const int mloc = waveM * 16 + r + 8 * h;
            atomicAdd(&s_w2[mloc], rw2[r]);
            atomicAdd(&s_r2[mloc], rr2[r]);
        }
        __syncthreads();

        // one thread per row: ball projection + hrelu collapse to one scalar
        if (tid < MT) {
            const float sw2 = s_w2[tid];
            const float sr2 = s_r2[tid];
            const float s1  = 1.f / (1.f + sqrtf(1.f + CCURV * sw2));
            const float nrm = fmaxf(s1 * sqrtf(sw2), EPSF);
            const float a1  = fminf(CS * nrm, 1.f - 1e-7f);
            const float ath = 0.5f * __logf((1.f + a1) / (1.f - a1)); // atanh
            const float g   = ath / (CS * nrm) * s1;                  // logmap
            const float vn  = fmaxf(g * sqrtf(sr2), EPSF);
            const float s   = tanhf(CS * vn) / (CS * vn) * g;         // expmap
            s_scale[tid] = s;
            out_ss[rowWG + tid] = s * s * sr2;   // next layer's sum(h^2)
        }
        __syncthreads();

        #pragma unroll
        for (int f = 0; f < NFRAG; ++f) {
            const int ncol = waveN * 16 * NFRAG + f * 16 + l16;
            #pragma unroll
            for (int r = 0; r < 8; ++r) {
                const int mloc = waveM * 16 + r + 8 * h;
                const float hv = s_scale[mloc] * fmaxf(acc[f][r], 0.f);
                out_bf16[(rowWG + mloc) * (long)FO_PAD + ncol] = (bf16_t)hv;
            }
        }
    } else {
        // fo==1 (padded to 16): only column 0 is real; sum(w^2)==w^2, no hrelu
        if (l16 == 0) {
            #pragma unroll
            for (int r = 0; r < 8; ++r) {
                const int mloc = waveM * 16 + r + 8 * h;
                const float w = acc[0][r];
                out_f32[rowWG + mloc] = w / (1.f + sqrtf(1.f + CCURV * w * w));
            }
        }
    }
}

// ---------------------------------------------------------------------------
extern "C" void kernel_launch(void* const* d_in, const int* in_sizes, int n_in,
                              void* d_out, int out_size, void* d_ws, size_t ws_size,
                              hipStream_t stream)
{
    const float* x  = (const float*)d_in[0];
    const float* z1 = (const float*)d_in[1];
    const float* b1 = (const float*)d_in[2];
    const float* z2 = (const float*)d_in[3];
    const float* b2 = (const float*)d_in[4];
    const float* z3 = (const float*)d_in[5];
    const float* b3 = (const float*)d_in[6];
    const float* z4 = (const float*)d_in[7];
    const float* b4 = (const float*)d_in[8];

    const int B = in_sizes[0] / 768;

    size_t off = 0;
    auto alloc = [&](size_t nbytes) -> void* {
        void* p = (char*)d_ws + off;
        off += (nbytes + 255) & ~(size_t)255;
        return p;
    };
    float*  ss0   = (float*)alloc((size_t)B * 4);
    float*  ssA   = (float*)alloc((size_t)B * 4);
    float*  ssB   = (float*)alloc((size_t)B * 4);
    bf16_t* z1T   = (bf16_t*)alloc((size_t)512 * 768 * 2);
    bf16_t* z2T   = (bf16_t*)alloc((size_t)256 * 512 * 2);
    bf16_t* z3T   = (bf16_t*)alloc((size_t)128 * 256 * 2);
    bf16_t* z4T   = (bf16_t*)alloc((size_t)16 * 128 * 2);
    float*  colp1 = (float*)alloc((size_t)512 * 3 * 4);
    float*  colp2 = (float*)alloc((size_t)256 * 3 * 4);
    float*  colp3 = (float*)alloc((size_t)128 * 3 * 4);
    float*  colp4 = (float*)alloc((size_t)16 * 3 * 4);
    bf16_t* h1    = (bf16_t*)alloc((size_t)B * 512 * 2);
    bf16_t* h2    = (bf16_t*)alloc((size_t)B * 256 * 2);
    bf16_t* h3    = (bf16_t*)alloc((size_t)B * 128 * 2);
    (void)ws_size;

    // --- prep: x row sumsq + bf16 transposed weights + column params ---
    row_sumsq_kernel<<<dim3((B * 32) / 256), dim3(256), 0, stream>>>(x, ss0, 768);
    prep_weights_kernel<768, 512, 512><<<dim3(8), dim3(64), 0, stream>>>(z1, b1, z1T, colp1);
    prep_weights_kernel<512, 256, 256><<<dim3(4), dim3(64), 0, stream>>>(z2, b2, z2T, colp2);
    prep_weights_kernel<256, 128, 128><<<dim3(2), dim3(64), 0, stream>>>(z3, b3, z3T, colp3);
    prep_weights_kernel<128, 1,   16><<<dim3(1), dim3(64), 0, stream>>>(z4, b4, z4T, colp4);

    // --- fused layers ---
    hlayer_kernel<768, 512, 2, 4, false, false>
        <<<dim3(B / 32), dim3(256), 0, stream>>>(x, nullptr, ss0, z1T, colp1, h1, ssA, nullptr);
    hlayer_kernel<512, 256, 2, 4, true, false>
        <<<dim3(B / 32), dim3(256), 0, stream>>>(nullptr, h1, ssA, z2T, colp2, h2, ssB, nullptr);
    hlayer_kernel<256, 128, 2, 4, true, false>
        <<<dim3(B / 32), dim3(256), 0, stream>>>(nullptr, h2, ssB, z3T, colp3, h3, ssA, nullptr);
    hlayer_kernel<128, 16, 8, 1, true, true>
        <<<dim3(B / 128), dim3(256), 0, stream>>>(nullptr, h3, ssA, z4T, colp4, nullptr, nullptr,
                                                  (float*)d_out);
}
